// ImgAttValMultiheadSelfAttention_40819369181624
// MI455X (gfx1250) — matrix operand order repaired and checked by reading, not compiled, not run on hardware
//
#include <hip/hip_runtime.h>
#include <hip/hip_bf16.h>

#define Bn  16
#define Pn  1024
#define En  1024
#define Hn  16
#define DHn 64

typedef __bf16 bf16;
typedef __attribute__((ext_vector_type(16))) bf16  v16bf;
typedef __attribute__((ext_vector_type(8)))  bf16  v8bf;
typedef __attribute__((ext_vector_type(4)))  bf16  v4bf;
typedef __attribute__((ext_vector_type(8)))  float v8f;

static __device__ __forceinline__ v16bf frag_from_halves(v8bf lo, v8bf hi) {
  v16bf r;
#pragma unroll
  for (int i = 0; i < 8; ++i) { r[i] = lo[i]; r[i + 8] = hi[i]; }
  return r;
}

// A-matrix fragment 16x32 (bf16) at (row0, k0) of a row-major matrix, leading dim lda.
// Per ISA: lanes 0-15 rows M, K base 0 (e<8: K=base+e; e>=8: K=16+base+e-8); lanes 16-31 K base 8.
static __device__ __forceinline__ v16bf load_frag_a(const bf16* __restrict__ m,
                                                    int row0, int lda, int k0) {
  const int lane = threadIdx.x & 31;
  const bf16* p = m + (size_t)(row0 + (lane & 15)) * lda + k0 + ((lane & 16) ? 8 : 0);
  return frag_from_halves(*(const v8bf*)p, *(const v8bf*)(p + 16));
}

// B-matrix fragment 32x16 (bf16) at (k0, n0), storage is B^T: bt[n][k], leading dim ldbt.
// Per ISA: lane n = col (mod 16), K base (lane<16 ? 0 : 8) (e<8: K=base+e; e>=8: K=16+base+e-8).
static __device__ __forceinline__ v16bf load_frag_bt(const bf16* __restrict__ bt,
                                                     int n0, int ldbt, int k0) {
  const int lane = threadIdx.x & 31;
  const bf16* p = bt + (size_t)(n0 + (lane & 15)) * ldbt + k0 + ((lane & 16) ? 8 : 0);
  return frag_from_halves(*(const v8bf*)p, *(const v8bf*)(p + 16));
}

static __device__ __forceinline__ v8f wmma_bf16(v16bf a, v16bf b, v8f c) {
  return __builtin_amdgcn_wmma_f32_16x16x32_bf16(false, a, false, b, (short)0, c,
                                                 false, false);
}

// ---------------------------------------------------------------------------
// Kernel 0a: fp32 -> bf16 bulk convert (4 elements/thread, b128 in / b64 out)
// ---------------------------------------------------------------------------
__global__ void __launch_bounds__(256)
cvt_f32_bf16_kernel(const float* __restrict__ in, bf16* __restrict__ out, int n4) {
  const int i = blockIdx.x * 256 + threadIdx.x;
  if (i < n4) {
    const float4 f = ((const float4*)in)[i];
    v4bf o;
    o[0] = (bf16)f.x; o[1] = (bf16)f.y; o[2] = (bf16)f.z; o[3] = (bf16)f.w;
    ((v4bf*)out)[i] = o;
  }
}

// ---------------------------------------------------------------------------
// Kernel 0b: w_att_val [H,DH,DH] f32 -> transposed bf16 wT[h][e][d] = w[h][d][e]
// ---------------------------------------------------------------------------
__global__ void __launch_bounds__(256)
watt_transpose_kernel(const float* __restrict__ w, bf16* __restrict__ wT) {
  const int h = blockIdx.x;
  for (int i = threadIdx.x; i < DHn * DHn; i += 256) {
    const int d = i >> 6, e = i & 63;
    wT[(size_t)h * DHn * DHn + (size_t)e * DHn + d] =
        (bf16)w[(size_t)h * DHn * DHn + (size_t)d * DHn + e];
  }
}

// ---------------------------------------------------------------------------
// Kernel 1: vx = x @ wv_w^T + wv_b ; stored transposed per head: vxT[B,H,DH,P]
// Each wave: 16 rows x 64 cols (one head), K=1024.
// Software-pipelined: prefetch k-step n+1 fragments, then issue step n WMMAs,
// so the backend can emit partial s_wait_loadcnt instead of full drains.
// ---------------------------------------------------------------------------
__global__ void __launch_bounds__(256)
gemm_vx_kernel(const bf16* __restrict__ xbf,   // [B*P, E]
               const bf16* __restrict__ wvbf,  // [E, E] == B^T (wv_w natural)
               const float* __restrict__ wvb,  // [E]
               bf16* __restrict__ vxT) {       // [B,H,DH,P]
  const int wave = threadIdx.x >> 5, lane = threadIdx.x & 31;
  const int wtile = blockIdx.x * 8 + wave;
  const int head = wtile & 15;
  const int r0 = (wtile >> 4) * 16;
  const int col = lane & 15, halfU = (lane >> 4) & 1;
  const int n0 = head * 64;

  v8f c0 = {}, c1 = {}, c2 = {}, c3 = {};
  v16bf a  = load_frag_a(xbf, r0, En, 0);
  v16bf b0 = load_frag_bt(wvbf, n0,      En, 0);
  v16bf b1 = load_frag_bt(wvbf, n0 + 16, En, 0);
  v16bf b2 = load_frag_bt(wvbf, n0 + 32, En, 0);
  v16bf b3 = load_frag_bt(wvbf, n0 + 48, En, 0);
  for (int k = 0; k < En - 32; k += 32) {
    const int kn = k + 32;
    const v16bf an  = load_frag_a(xbf, r0, En, kn);
    const v16bf bn0 = load_frag_bt(wvbf, n0,      En, kn);
    const v16bf bn1 = load_frag_bt(wvbf, n0 + 16, En, kn);
    const v16bf bn2 = load_frag_bt(wvbf, n0 + 32, En, kn);
    const v16bf bn3 = load_frag_bt(wvbf, n0 + 48, En, kn);
    c0 = wmma_bf16(a, b0, c0);
    c1 = wmma_bf16(a, b1, c1);
    c2 = wmma_bf16(a, b2, c2);
    c3 = wmma_bf16(a, b3, c3);
    a = an; b0 = bn0; b1 = bn1; b2 = bn2; b3 = bn3;
  }
  c0 = wmma_bf16(a, b0, c0);
  c1 = wmma_bf16(a, b1, c1);
  c2 = wmma_bf16(a, b2, c2);
  c3 = wmma_bf16(a, b3, c3);

  const int b  = r0 >> 10;        // r0 / P
  const int p0 = r0 & (Pn - 1);

#define VX_STORE(cN, nb)                                                          \
  {                                                                               \
    const float bias = wvb[n0 + (nb) * 16 + col];                                 \
    bf16* dst = vxT + (((size_t)(b * Hn + head)) * DHn + (nb) * 16 + col) * Pn +  \
                p0 + halfU * 8;                                                   \
    v8bf pk;                                                                      \
    _Pragma("unroll") for (int v = 0; v < 8; ++v) pk[v] = (bf16)(cN[v] + bias);   \
    *(v8bf*)dst = pk;                                                             \
  }
  VX_STORE(c0, 0) VX_STORE(c1, 1) VX_STORE(c2, 2) VX_STORE(c3, 3)
#undef VX_STORE
}

// ---------------------------------------------------------------------------
// Kernel 2: Q[b][h][p][d] = x_h @ w_att_val[h]  (K=64: straight-line, all loads
// first, then 8 WMMAs, letting the scheduler interleave)
// ---------------------------------------------------------------------------
__global__ void __launch_bounds__(256)
gemm_q_kernel(const bf16* __restrict__ xbf,  // [B*P, E]
              const bf16* __restrict__ wT,   // [H][e][d] (B^T per head)
              bf16* __restrict__ qbf) {      // [B,H,P,DH]
  const int wave = threadIdx.x >> 5, lane = threadIdx.x & 31;
  const int wtile = blockIdx.x * 8 + wave;
  const int head = wtile & 15;
  const int r0 = (wtile >> 4) * 16;
  const int col = lane & 15, halfU = (lane >> 4) & 1;

  const bf16* abase = xbf + head * DHn;                // head's column slice
  const bf16* bt = wT + (size_t)head * DHn * DHn;

  const v16bf a0  = load_frag_a(abase, r0, En, 0);
  const v16bf a1  = load_frag_a(abase, r0, En, 32);
  const v16bf b00 = load_frag_bt(bt,  0, DHn, 0);
  const v16bf b01 = load_frag_bt(bt, 16, DHn, 0);
  const v16bf b02 = load_frag_bt(bt, 32, DHn, 0);
  const v16bf b03 = load_frag_bt(bt, 48, DHn, 0);
  const v16bf b10 = load_frag_bt(bt,  0, DHn, 32);
  const v16bf b11 = load_frag_bt(bt, 16, DHn, 32);
  const v16bf b12 = load_frag_bt(bt, 32, DHn, 32);
  const v16bf b13 = load_frag_bt(bt, 48, DHn, 32);

  v8f c0 = {}, c1 = {}, c2 = {}, c3 = {};
  c0 = wmma_bf16(a0, b00, c0);
  c1 = wmma_bf16(a0, b01, c1);
  c2 = wmma_bf16(a0, b02, c2);
  c3 = wmma_bf16(a0, b03, c3);
  c0 = wmma_bf16(a1, b10, c0);
  c1 = wmma_bf16(a1, b11, c1);
  c2 = wmma_bf16(a1, b12, c2);
  c3 = wmma_bf16(a1, b13, c3);

  const int b = r0 >> 10, p0 = r0 & (Pn - 1);
  bf16* dst = qbf + (((size_t)(b * Hn + head)) * Pn + p0) * DHn;
#pragma unroll
  for (int v = 0; v < 8; ++v) {
    const int row = halfU * 8 + v;
    dst[(size_t)row * DHn + col]      = (bf16)c0[v];
    dst[(size_t)row * DHn + 16 + col] = (bf16)c1[v];
    dst[(size_t)row * DHn + 32 + col] = (bf16)c2[v];
    dst[(size_t)row * DHn + 48 + col] = (bf16)c3[v];
  }
}

// ---------------------------------------------------------------------------
// Kernel 3: streaming-softmax attention. One wave = 16 query rows, full DH=64.
// Key tiles of 32. V fragments and bias scalars are loaded BEFORE the S-WMMAs:
// loads return in order, so S-WMMA waits only past K-frags, softmax waits only
// past bias, and V latency is hidden under the exp/shuffle VALU work.
// ---------------------------------------------------------------------------
__global__ void __launch_bounds__(256)
flash_attn_kernel(const bf16* __restrict__ qbf,   // [B,H,P,DH]
                  const bf16* __restrict__ xbf,   // [B,P,E]  (keys, head slice)
                  const bf16* __restrict__ vxT,   // [B,H,DH,P]
                  const float* __restrict__ bias, // [H,P,P]
                  bf16* __restrict__ obf) {       // [B,P,E]
  __shared__ bf16 lds_p[8 * 16 * 32];
  const int wave = threadIdx.x >> 5;
  const int lane = threadIdx.x & 31;
  bf16* ptile = lds_p + wave * (16 * 32);

  const int mblk = blockIdx.x & 7;
  const int bh   = blockIdx.x >> 3;
  const int h    = bh & (Hn - 1);
  const int b    = bh >> 4;
  const int q0   = mblk * 128 + wave * 16;

  const int col   = lane & 15;
  const int halfU = (lane >> 4) & 1;

  const bf16*  qmat  = qbf + ((size_t)(b * Hn + h) * Pn) * DHn;
  const bf16*  kmat  = xbf + (size_t)b * Pn * En + h * DHn;  // K[p][d], ld=E
  const bf16*  vtmat = vxT + ((size_t)(b * Hn + h) * DHn) * Pn;
  const float* brow  = bias + (size_t)h * Pn * Pn + (size_t)q0 * Pn;

  const v16bf qf0 = load_frag_a(qmat, q0, DHn, 0);
  const v16bf qf1 = load_frag_a(qmat, q0, DHn, 32);

  v8f o0 = {}, o1 = {}, o2 = {}, o3 = {};
  float m8[8], l8[8];
#pragma unroll
  for (int v = 0; v < 8; ++v) { m8[v] = -3.0e38f; l8[v] = 0.0f; }

  const float scale = 0.125f;  // 1/sqrt(64)

  for (int kt = 0; kt < Pn; kt += 32) {
    // --- issue all global loads for this tile up front (in-order returns) ---
    const v16bf k00 = load_frag_bt(kmat, kt,      En, 0);
    const v16bf k01 = load_frag_bt(kmat, kt,      En, 32);
    const v16bf k10 = load_frag_bt(kmat, kt + 16, En, 0);
    const v16bf k11 = load_frag_bt(kmat, kt + 16, En, 32);
    float bv0[8], bv1[8];
#pragma unroll
    for (int v = 0; v < 8; ++v) {
      const int row = halfU * 8 + v;
      bv0[v] = brow[(size_t)row * Pn + kt + col];
      bv1[v] = brow[(size_t)row * Pn + kt + 16 + col];
    }
    const v16bf vb0 = load_frag_bt(vtmat,  0, Pn, kt);
    const v16bf vb1 = load_frag_bt(vtmat, 16, Pn, kt);
    const v16bf vb2 = load_frag_bt(vtmat, 32, Pn, kt);
    const v16bf vb3 = load_frag_bt(vtmat, 48, Pn, kt);

    // --- S = Q K^T ---
    v8f s0 = {}, s1 = {};
    s0 = wmma_bf16(qf0, k00, s0);
    s0 = wmma_bf16(qf1, k01, s0);
    s1 = wmma_bf16(qf0, k10, s1);
    s1 = wmma_bf16(qf1, k11, s1);

    // --- online softmax over the 32-key tile ---
#pragma unroll
    for (int v = 0; v < 8; ++v) {
      const int row = halfU * 8 + v;
      const float t0 = s0[v] * scale + bv0[v];
      const float t1 = s1[v] * scale + bv1[v];
      float rm = fmaxf(t0, t1);
      rm = fmaxf(rm, __shfl_xor(rm, 1, 32));   // masks <16 stay inside 16-lane half
      rm = fmaxf(rm, __shfl_xor(rm, 2, 32));
      rm = fmaxf(rm, __shfl_xor(rm, 4, 32));
      rm = fmaxf(rm, __shfl_xor(rm, 8, 32));
      const float mnew  = fmaxf(m8[v], rm);
      const float alpha = __expf(m8[v] - mnew);
      const float e0 = __expf(t0 - mnew);
      const float e1 = __expf(t1 - mnew);
      float rs = e0 + e1;
      rs += __shfl_xor(rs, 1, 32);
      rs += __shfl_xor(rs, 2, 32);
      rs += __shfl_xor(rs, 4, 32);
      rs += __shfl_xor(rs, 8, 32);
      m8[v] = mnew;
      l8[v] = l8[v] * alpha + rs;
      o0[v] *= alpha; o1[v] *= alpha; o2[v] *= alpha; o3[v] *= alpha;
      ptile[row * 32 + col]      = (bf16)e0;   // C-layout -> [16][32] bf16 tile
      ptile[row * 32 + 16 + col] = (bf16)e1;
    }
    // Same-wave cross-lane LDS RAW: force dscnt drain + forbid reordering.
    asm volatile("s_wait_dscnt 0x0" ::: "memory");

    // --- O += P V ---
    {
      const bf16* pp = ptile + (lane & 15) * 32 + ((lane & 16) ? 8 : 0);
      const v16bf pa = frag_from_halves(*(const v8bf*)pp, *(const v8bf*)(pp + 16));
      o0 = wmma_bf16(pa, vb0, o0);
      o1 = wmma_bf16(pa, vb1, o1);
      o2 = wmma_bf16(pa, vb2, o2);
      o3 = wmma_bf16(pa, vb3, o3);
    }
  }

  bf16* orow = obf + ((size_t)b * Pn + q0) * En + h * DHn;  // merged-head layout
#pragma unroll
  for (int v = 0; v < 8; ++v) {
    const int row = halfU * 8 + v;
    const float inv = 1.0f / l8[v];
    orow[(size_t)row * En + col]      = (bf16)(o0[v] * inv);
    orow[(size_t)row * En + 16 + col] = (bf16)(o1[v] * inv);
    orow[(size_t)row * En + 32 + col] = (bf16)(o2[v] * inv);
    orow[(size_t)row * En + 48 + col] = (bf16)(o3[v] * inv);
  }
}

// ---------------------------------------------------------------------------
// Kernel 4: out = O @ out_w^T + out_b  -> fp32 d_out  (pipelined like kernel 1)
// ---------------------------------------------------------------------------
__global__ void __launch_bounds__(256)
gemm_out_kernel(const bf16* __restrict__ obf,   // [B*P, E]
                const bf16* __restrict__ owbf,  // [E, E] == B^T (out_w natural)
                const float* __restrict__ ob,   // [E]
                float* __restrict__ out) {      // [B*P, E] fp32
  const int wave = threadIdx.x >> 5, lane = threadIdx.x & 31;
  const int wtile = blockIdx.x * 8 + wave;
  const int cblk = wtile & 15;
  const int r0 = (wtile >> 4) * 16;
  const int col = lane & 15, halfU = (lane >> 4) & 1;
  const int n0 = cblk * 64;

  v8f c0 = {}, c1 = {}, c2 = {}, c3 = {};
  v16bf a  = load_frag_a(obf, r0, En, 0);
  v16bf b0 = load_frag_bt(owbf, n0,      En, 0);
  v16bf b1 = load_frag_bt(owbf, n0 + 16, En, 0);
  v16bf b2 = load_frag_bt(owbf, n0 + 32, En, 0);
  v16bf b3 = load_frag_bt(owbf, n0 + 48, En, 0);
  for (int k = 0; k < En - 32; k += 32) {
    const int kn = k + 32;
    const v16bf an  = load_frag_a(obf, r0, En, kn);
    const v16bf bn0 = load_frag_bt(owbf, n0,      En, kn);
    const v16bf bn1 = load_frag_bt(owbf, n0 + 16, En, kn);
    const v16bf bn2 = load_frag_bt(owbf, n0 + 32, En, kn);
    const v16bf bn3 = load_frag_bt(owbf, n0 + 48, En, kn);
    c0 = wmma_bf16(a, b0, c0);
    c1 = wmma_bf16(a, b1, c1);
    c2 = wmma_bf16(a, b2, c2);
    c3 = wmma_bf16(a, b3, c3);
    a = an; b0 = bn0; b1 = bn1; b2 = bn2; b3 = bn3;
  }
  c0 = wmma_bf16(a, b0, c0);
  c1 = wmma_bf16(a, b1, c1);
  c2 = wmma_bf16(a, b2, c2);
  c3 = wmma_bf16(a, b3, c3);

  const float bb0 = ob[n0 + col];
  const float bb1 = ob[n0 + 16 + col];
  const float bb2 = ob[n0 + 32 + col];
  const float bb3 = ob[n0 + 48 + col];
#pragma unroll
  for (int v = 0; v < 8; ++v) {
    const size_t row = (size_t)(r0 + halfU * 8 + v);
    out[row * En + n0 + col]      = c0[v] + bb0;
    out[row * En + n0 + 16 + col] = c1[v] + bb1;
    out[row * En + n0 + 32 + col] = c2[v] + bb2;
    out[row * En + n0 + 48 + col] = c3[v] + bb3;
  }
}

// ---------------------------------------------------------------------------
extern "C" void kernel_launch(void* const* d_in, const int* in_sizes, int n_in,
                              void* d_out, int out_size, void* d_ws, size_t ws_size,
                              hipStream_t stream) {
  const float* x    = (const float*)d_in[0];  // [B,P,E]
  const float* watt = (const float*)d_in[1];  // [H,DH,DH]
  const float* bias = (const float*)d_in[2];  // [H,P,P]
  const float* wv_w = (const float*)d_in[3];  // [E,E]
  const float* wv_b = (const float*)d_in[4];  // [E]
  const float* ow   = (const float*)d_in[5];  // [E,E]
  const float* obv  = (const float*)d_in[6];  // [E]
  float* out = (float*)d_out;
  (void)in_sizes; (void)n_in; (void)out_size; (void)ws_size;

  char* ws = (char*)d_ws;
  size_t off = 0;
  auto take = [&](size_t bytes) -> char* {
    char* p = ws + off;
    off += (bytes + 255) & ~(size_t)255;
    return p;
  };
  bf16* xbf  = (bf16*)take((size_t)Bn * Pn * En * 2);        // 32 MB
  bf16* wvbf = (bf16*)take((size_t)En * En * 2);             // 2 MB
  bf16* owbf = (bf16*)take((size_t)En * En * 2);             // 2 MB
  bf16* wTbf = (bf16*)take((size_t)Hn * DHn * DHn * 2);      // 128 KB
  bf16* qbf  = (bf16*)take((size_t)Bn * Hn * Pn * DHn * 2);  // 32 MB
  bf16* vxT  = (bf16*)take((size_t)Bn * Hn * DHn * Pn * 2);  // 32 MB
  bf16* obf  = (bf16*)take((size_t)Bn * Pn * En * 2);        // 32 MB

  const int nx4 = Bn * Pn * En / 4;
  const int nw4 = En * En / 4;
  cvt_f32_bf16_kernel<<<(nx4 + 255) / 256, 256, 0, stream>>>(x, xbf, nx4);
  cvt_f32_bf16_kernel<<<(nw4 + 255) / 256, 256, 0, stream>>>(wv_w, wvbf, nw4);
  cvt_f32_bf16_kernel<<<(nw4 + 255) / 256, 256, 0, stream>>>(ow, owbf, nw4);
  watt_transpose_kernel<<<Hn, 256, 0, stream>>>(watt, wTbf);

  const int gemm_blocks = (Bn * Pn / 16) * (En / 64) / 8;  // 2048 WGs, 8 waves each
  gemm_vx_kernel<<<gemm_blocks, 256, 0, stream>>>(xbf, wvbf, wv_b, vxT);
  gemm_q_kernel<<<gemm_blocks, 256, 0, stream>>>(xbf, wTbf, qbf);
  flash_attn_kernel<<<Bn * Hn * (Pn / 128), 256, 0, stream>>>(qbf, xbf, vxT, bias, obf);
  gemm_out_kernel<<<gemm_blocks, 256, 0, stream>>>(obf, owbf, obv, out);
}
